// TemporalModule_77764677861990
// MI455X (gfx1250) — matrix operand order repaired and checked by reading, not compiled, not run on hardware
//
#include <hip/hip_runtime.h>

// ---- problem constants (fixed by setup_inputs) ----
#define HEADS 8
#define DH 40
#define F 16
#define C 320
#define D 4096
#define G 4              // sequences per block
#define XSTRIDE 328      // padded row stride (halfs) for F x C LDS tiles
#define VTS 16           // Vt row stride (frames per dim row)
#define NTHREADS 512
#define NWAVES (NTHREADS / 32)
#define WSZ (C * C)      // elements per weight matrix
#define KCHUNKS (C / 32) // 10 (even)

typedef _Float16 v8h  __attribute__((ext_vector_type(8)));
typedef _Float16 v16h __attribute__((ext_vector_type(16)));
typedef float    v8f  __attribute__((ext_vector_type(8)));

__device__ __forceinline__ v16h cat8(v8h lo, v8h hi) {
  return __builtin_shufflevector(lo, hi, 0, 1, 2, 3, 4, 5, 6, 7,
                                 8, 9, 10, 11, 12, 13, 14, 15);
}

__device__ __forceinline__ v8f wmma16(v16h a, v16h b, v8f c) {
  // D = A(16x32 f16) * B(32x16 f16) + C(16x16 f32)
  return __builtin_amdgcn_wmma_f32_16x16x32_f16(false, a, false, b,
                                                (short)0, c, false, false);
}

// A fragment (16x32 f16, ISA 7.12.2): lane<16 -> row=lane, k in {k0..k0+7, k0+16..k0+23};
// lane>=16 -> row=lane-16, k in {k0+8..k0+15, k0+24..k0+31}.
__device__ __forceinline__ v16h load_a(const _Float16* row0, int stride, int k0, int lane) {
  const _Float16* p = row0 + (lane & 15) * stride + k0 + ((lane & 16) ? 8 : 0);
  v8h lo = *(const v8h*)p;
  v8h hi = *(const v8h*)(p + 16);
  return cat8(lo, hi);
}

// B fragment (32x16 f16) with B[k][n] = W[n][k], W row-major [N][K]:
// lane<16 -> col n0+lane, k in {k0..k0+7, k0+16..k0+23}; lane>=16 -> col n0+lane-16, k in {k0+8..k0+15, k0+24..k0+31}.
__device__ __forceinline__ v16h load_b(const _Float16* w, int stride, int n0, int k0, int lane) {
  const _Float16* p = w + (n0 + (lane & 15)) * stride + k0 + ((lane & 16) ? 8 : 0);
  v8h lo = *(const v8h*)p;
  v8h hi = *(const v8h*)(p + 16);
  return cat8(lo, hi);
}

// ---- kernel 0: weights fp32 -> f16 in workspace ----
__global__ void convert_weights_f16(const float* __restrict__ Wq,
                                    const float* __restrict__ Wk,
                                    const float* __restrict__ Wv,
                                    const float* __restrict__ Wo,
                                    _Float16* __restrict__ w16) {
  int i = blockIdx.x * blockDim.x + threadIdx.x;
  if (i < WSZ) {
    w16[0 * WSZ + i] = (_Float16)Wq[i];
    w16[1 * WSZ + i] = (_Float16)Wk[i];
    w16[2 * WSZ + i] = (_Float16)Wv[i];
    w16[3 * WSZ + i] = (_Float16)Wo[i];
  }
}

// ---- kernel 1: fused temporal attention ----
__global__ __launch_bounds__(NTHREADS)
void temporal_attn_kernel(const float* __restrict__ hidden,
                          const _Float16* __restrict__ w16,
                          const float* __restrict__ bo,
                          float* __restrict__ out) {
  __shared__ __align__(16) _Float16 XO[G][F][XSTRIDE];  // x, later attn output
  __shared__ __align__(16) _Float16 Qs[G][F][XSTRIDE];
  __shared__ __align__(16) _Float16 Ks[G][F][XSTRIDE];
  __shared__ __align__(16) _Float16 Vt[G][C][VTS];      // V transposed: [dim][frame]

  const int lane  = threadIdx.x & 31;
  const int wave  = threadIdx.x >> 5;
  const int col   = lane & 15;
  const int rbase = (lane & 16) ? 8 : 0;
  const int seq0  = blockIdx.x * G;

  // ---- Phase 1: gather x (strided frames), add APE, convert to f16 ----
  for (int idx = threadIdx.x; idx < G * F * C; idx += NTHREADS) {
    int g   = idx / (F * C);
    int rem = idx % (F * C);
    int fr  = rem / C;
    int ch  = rem % C;
    int s   = seq0 + g;
    int bi  = s >> 12;          // s / D
    int di  = s & (D - 1);      // s % D
    float v = hidden[(size_t)((bi * F + fr) * D + di) * C + ch];
    float dv  = __expf((float)(ch & ~1) * (-0.02878231366f)); // -ln(1e4)/320
    float arg = (float)fr * dv;
    v += (ch & 1) ? __cosf(arg) : __sinf(arg);
    XO[g][fr][ch] = (_Float16)v;
  }
  __syncthreads();

  // ---- Phase 2: Q, K, V projections (bias-free); V stored transposed ----
  for (int task = wave; task < 3 * (C / 16); task += NWAVES) {
    int mat = task / (C / 16);
    int nt  = task % (C / 16);
    const _Float16* W = w16 + mat * WSZ;
    v8f acc[G] = {};
    // two independently-loaded B buffers: no VALU rotation copies -> no WMMA
    // WAR-hazard v_nops; each buffer is written only by its own global_load.
    v16h bf0 = load_b(W, C, nt * 16, 0, lane);
#pragma unroll 1
    for (int kc = 0; kc < KCHUNKS; kc += 2) {
      v16h bf1 = load_b(W, C, nt * 16, (kc + 1) * 32, lane);
#pragma unroll
      for (int g = 0; g < G; ++g)
        acc[g] = wmma16(load_a(&XO[g][0][0], XSTRIDE, kc * 32, lane), bf0, acc[g]);
      bf0 = load_b(W, C, nt * 16, (kc + 2 < KCHUNKS ? kc + 2 : 0) * 32, lane);
#pragma unroll
      for (int g = 0; g < G; ++g)
        acc[g] = wmma16(load_a(&XO[g][0][0], XSTRIDE, (kc + 1) * 32, lane), bf1, acc[g]);
    }
#pragma unroll
    for (int g = 0; g < G; ++g) {
      if (mat == 2) {
#pragma unroll
        for (int r = 0; r < 8; ++r)
          Vt[g][nt * 16 + col][rbase + r] = (_Float16)acc[g][r];
      } else {
        _Float16* dst = (mat == 0) ? &Qs[g][0][0] : &Ks[g][0][0];
#pragma unroll
        for (int r = 0; r < 8; ++r)
          dst[(rbase + r) * XSTRIDE + nt * 16 + col] = (_Float16)acc[g][r];
      }
    }
  }
  __syncthreads();

  // ---- Phase 3: per-(seq, head) causal attention ----
  const v8h z8 = {};
  for (int task = wave; task < G * HEADS; task += NWAVES) {
    int g = task >> 3;
    int h = task & 7;
    const _Float16* Qg = &Qs[g][0][0];
    const _Float16* Kg = &Ks[g][0][0];
    _Float16*       Og = &XO[g][0][0];

    // scores = Q_h @ K_h^T (dh=40 padded to K=64 -> 2 WMMAs)
    v8f sc = {};
    {
      v16h a = load_a(Qg, XSTRIDE, h * DH, lane);
      v16h b = load_b(Kg, XSTRIDE, 0, h * DH, lane);
      sc = wmma16(a, b, sc);
    }
    {
      v8h alo = (lane < 16) ? *(const v8h*)(Qg + col * XSTRIDE + h * DH + 32) : z8;
      v8h blo = (lane < 16) ? *(const v8h*)(Kg + col * XSTRIDE + h * DH + 32) : z8;
      sc = wmma16(cat8(alo, z8), cat8(blo, z8), sc);
    }

    // scale + causal mask + softmax per row (16-lane butterfly reductions)
    float pr[8];
#pragma unroll
    for (int r = 0; r < 8; ++r) {
      int row = rbase + r;                 // query frame
      bool valid = (col <= row);           // causal: key <= query
      float sv = valid ? sc[r] * 0.15811388f : -1e30f; // 1/sqrt(40)
      float m = sv;
      m = fmaxf(m, __shfl_xor(m, 1));
      m = fmaxf(m, __shfl_xor(m, 2));
      m = fmaxf(m, __shfl_xor(m, 4));
      m = fmaxf(m, __shfl_xor(m, 8));
      float e = valid ? __expf(sv - m) : 0.0f;
      float t = e;
      t += __shfl_xor(t, 1);
      t += __shfl_xor(t, 2);
      t += __shfl_xor(t, 4);
      t += __shfl_xor(t, 8);
      pr[r] = e / t;
    }
    // stash P (f16) into this head's private slice of the output buffer
#pragma unroll
    for (int r = 0; r < 8; ++r)
      Og[(rbase + r) * XSTRIDE + h * DH + col] = (_Float16)pr[r];

    // out_h = P (16x16, K padded to 32) @ V_h (16x40)
    v8h alo = *(const v8h*)(Og + col * XSTRIDE + h * DH + ((lane & 16) ? 8 : 0));
    v16h a = cat8(alo, z8);
    const _Float16* Vg = &Vt[g][0][0];
#pragma unroll
    for (int ntile = 0; ntile < 3; ++ntile) {
      int nc  = ntile * 16 + col;               // dim within head (pad to 48)
      int dim = h * DH + (nc < DH ? nc : 0);
      v16h b;
      {
        v8h blo = *(const v8h*)(Vg + dim * VTS + ((lane & 16) ? 8 : 0));
        b = cat8(blo, z8);                      // k = frames 8h..8h+7, then k = 16..31 padding
      }
      v8f o = {};
      o = wmma16(a, b, o);
      if (nc < DH) {
#pragma unroll
        for (int r = 0; r < 8; ++r)
          Og[(rbase + r) * XSTRIDE + h * DH + nc] = (_Float16)o[r];
      }
    }
  }
  __syncthreads();

  // ---- Phase 4: output projection + bias, scatter to (b f) d c ----
  const _Float16* Wo16 = w16 + 3 * WSZ;
  for (int nt = wave; nt < C / 16; nt += NWAVES) {
    float bias = bo[nt * 16 + col];
    v8f acc[G] = {};
    v16h bf0 = load_b(Wo16, C, nt * 16, 0, lane);
#pragma unroll 1
    for (int kc = 0; kc < KCHUNKS; kc += 2) {
      v16h bf1 = load_b(Wo16, C, nt * 16, (kc + 1) * 32, lane);
#pragma unroll
      for (int g = 0; g < G; ++g)
        acc[g] = wmma16(load_a(&XO[g][0][0], XSTRIDE, kc * 32, lane), bf0, acc[g]);
      bf0 = load_b(Wo16, C, nt * 16, (kc + 2 < KCHUNKS ? kc + 2 : 0) * 32, lane);
#pragma unroll
      for (int g = 0; g < G; ++g)
        acc[g] = wmma16(load_a(&XO[g][0][0], XSTRIDE, (kc + 1) * 32, lane), bf1, acc[g]);
    }
#pragma unroll
    for (int g = 0; g < G; ++g) {
      int s  = seq0 + g;
      int bi = s >> 12;
      int di = s & (D - 1);
      int n  = nt * 16 + col;
#pragma unroll
      for (int r = 0; r < 8; ++r) {
        int row = rbase + r;
        out[(size_t)((bi * F + row) * D + di) * C + n] = acc[g][r] + bias;
      }
    }
  }
}

extern "C" void kernel_launch(void* const* d_in, const int* in_sizes, int n_in,
                              void* d_out, int out_size, void* d_ws, size_t ws_size,
                              hipStream_t stream) {
  const float* hidden = (const float*)d_in[0];
  const float* Wq     = (const float*)d_in[1];
  const float* Wk     = (const float*)d_in[2];
  const float* Wv     = (const float*)d_in[3];
  const float* Wo     = (const float*)d_in[4];
  const float* bo     = (const float*)d_in[5];
  float*       out    = (float*)d_out;
  _Float16*    w16    = (_Float16*)d_ws;   // 4 * 320*320 * 2B = 819200 B

  convert_weights_f16<<<(WSZ + 255) / 256, 256, 0, stream>>>(Wq, Wk, Wv, Wo, w16);

  const int nblocks = (2 * D) / G;          // 8192 sequences / G
  temporal_attn_kernel<<<nblocks, NTHREADS, 0, stream>>>(hidden, w16, bo, out);
}
